// Transformer_4982162063789
// MI455X (gfx1250) — matrix operand order repaired and checked by reading, not compiled, not run on hardware
//
#include <hip/hip_runtime.h>
#include <hip/hip_bf16.h>

typedef __attribute__((ext_vector_type(16))) __bf16 v16bf;
typedef __attribute__((ext_vector_type(8)))  __bf16 v8bf;
typedef __attribute__((ext_vector_type(8)))  float  v8f;

#define VOCAB  32000
#define DMODEL 1024
#define BATCH  4
#define SEQ    2048

// ---------------- conversion / gather ----------------

__global__ void cvt_f32_bf16(const float* __restrict__ src, __bf16* __restrict__ dst, long n) {
  long i = (long)blockIdx.x * blockDim.x + threadIdx.x;
  long stride = (long)gridDim.x * blockDim.x;
  for (; i < n; i += stride) dst[i] = (__bf16)src[i];
}

__global__ void embed_gather(const int* __restrict__ idx, const float* __restrict__ embed,
                             __bf16* __restrict__ xbf) {
  int row = blockIdx.x;                 // 0 .. B*S-1
  long src = (long)idx[row] * DMODEL;
  long dst = (long)row * DMODEL;
  for (int d = threadIdx.x; d < DMODEL; d += 256)
    xbf[dst + d] = (__bf16)embed[src + d];
}

// ---------------- WMMA GEMM: C = A * B^T ----------------
// A: [M,K] rows (lda), B: [N,K] rows (ldb), C: [M,N] (ldc). Per-batch strides in elements.
// Block tile 128x128, 8 wave32 waves (4 M-strips x 2 N-strips), 32x64 per wave
// (2x4 WMMA 16x16 tiles -> 8 WMMAs per 6 LDS fragment loads per K-step of 32).

#define BM 128
#define BN 128
#define BK 32
#define S_STRIDE 40   // bf16 elems per LDS row (32 + 8 pad) -> 80B, 16B-aligned chunks

enum { EPI_F32 = 0, EPI_BF16 = 1, EPI_BF16_T = 2, EPI_SCORES = 3, EPI_BIAS = 4 };

// A fragment (16x32 bf16): lane l holds row (l&15); elem e -> K = (e>>3)*16 + (l>>4)*8 + (e&7)
__device__ __forceinline__ v16bf frag_ld_A(const __bf16* s, int rowbase, int lane) {
  int g = lane >> 4, r = lane & 15;
  const __bf16* p = s + (rowbase + r) * S_STRIDE;
  union { v16bf v; v8bf h[2]; } u;
  u.h[0] = *(const v8bf*)(p + g * 8);        // K = g*8 + 0..7
  u.h[1] = *(const v8bf*)(p + 16 + g * 8);   // K = 16 + g*8 + 0..7
  return u.v;
}

// B fragment (32x16 bf16): lane l holds col (l&15); elem e -> K = (e>>3)*16 + (l>>4)*8 + (e&7)
__device__ __forceinline__ v16bf frag_ld_B(const __bf16* s, int rowbase, int lane) {
  int g = lane >> 4, r = lane & 15;
  const __bf16* p = s + (rowbase + r) * S_STRIDE;
  union { v16bf v; v8bf h[2]; } u;
  u.h[0] = *(const v8bf*)(p + g * 8);
  u.h[1] = *(const v8bf*)(p + 16 + g * 8);
  return u.v;
}

template <int MODE>
__global__ __launch_bounds__(256)
void gemm_bf16(const __bf16* __restrict__ A, long strideA, int lda,
               const __bf16* __restrict__ Bm, long strideB, int ldb,
               void* __restrict__ Cout, long strideC, int ldc,
               int K, const float* __restrict__ bias, float scale) {
  __shared__ __bf16 sA[BM * S_STRIDE];
  __shared__ __bf16 sB[BN * S_STRIDE];

  const int tid  = threadIdx.x;
  const int lane = tid & 31;
  const int wave = tid >> 5;
  const int wm   = wave >> 1;   // 0..3 -> 32-row strip
  const int wn   = wave & 1;    // 0..1 -> 64-col strip
  const int bM   = blockIdx.y * BM;
  const int bN   = blockIdx.x * BN;
  const int z    = blockIdx.z;

  const __bf16* gA = A  + (long)z * strideA + (long)bM * lda;
  const __bf16* gB = Bm + (long)z * strideB + (long)bN * ldb;

  // cooperative-load indices: 512 A-chunks + 512 B-chunks of 8 bf16 (16B), 256 threads
  const int row0 = tid >> 2;                 // chunks tid      (rows   0..63)
  const int row1 = (tid + 256) >> 2;         // chunks tid+256  (rows  64..127)
  const int cc   = (tid & 3) * 8;

  v8f acc[2][4] = {};

  for (int k0 = 0; k0 < K; k0 += BK) {
    *(v8bf*)&sA[row0 * S_STRIDE + cc] = *(const v8bf*)&gA[(long)row0 * lda + k0 + cc];
    *(v8bf*)&sA[row1 * S_STRIDE + cc] = *(const v8bf*)&gA[(long)row1 * lda + k0 + cc];
    *(v8bf*)&sB[row0 * S_STRIDE + cc] = *(const v8bf*)&gB[(long)row0 * ldb + k0 + cc];
    *(v8bf*)&sB[row1 * S_STRIDE + cc] = *(const v8bf*)&gB[(long)row1 * ldb + k0 + cc];
    if (k0 + BK < K) {   // global_prefetch_b8, keep near (L2-resident weights)
      __builtin_prefetch(&gA[(long)row0 * lda + k0 + BK + cc], 0, 3);
      __builtin_prefetch(&gB[(long)row0 * ldb + k0 + BK + cc], 0, 3);
    }
    __syncthreads();

    v16bf a0 = frag_ld_A(sA, wm * 32,      lane);
    v16bf a1 = frag_ld_A(sA, wm * 32 + 16, lane);
    v16bf b0 = frag_ld_B(sB, wn * 64,      lane);
    v16bf b1 = frag_ld_B(sB, wn * 64 + 16, lane);
    v16bf b2 = frag_ld_B(sB, wn * 64 + 32, lane);
    v16bf b3 = frag_ld_B(sB, wn * 64 + 48, lane);

    acc[0][0] = __builtin_amdgcn_wmma_f32_16x16x32_bf16(false, a0, false, b0, (short)0, acc[0][0], false, false);
    acc[0][1] = __builtin_amdgcn_wmma_f32_16x16x32_bf16(false, a0, false, b1, (short)0, acc[0][1], false, false);
    acc[0][2] = __builtin_amdgcn_wmma_f32_16x16x32_bf16(false, a0, false, b2, (short)0, acc[0][2], false, false);
    acc[0][3] = __builtin_amdgcn_wmma_f32_16x16x32_bf16(false, a0, false, b3, (short)0, acc[0][3], false, false);
    acc[1][0] = __builtin_amdgcn_wmma_f32_16x16x32_bf16(false, a1, false, b0, (short)0, acc[1][0], false, false);
    acc[1][1] = __builtin_amdgcn_wmma_f32_16x16x32_bf16(false, a1, false, b1, (short)0, acc[1][1], false, false);
    acc[1][2] = __builtin_amdgcn_wmma_f32_16x16x32_bf16(false, a1, false, b2, (short)0, acc[1][2], false, false);
    acc[1][3] = __builtin_amdgcn_wmma_f32_16x16x32_bf16(false, a1, false, b3, (short)0, acc[1][3], false, false);
    __syncthreads();
  }

  // epilogue: C/D layout -> VGPR r holds row r + 8*(lane>>4), col lane&15
  const int g  = lane >> 4;
  const int nl = lane & 15;
  const long cbase = (long)z * strideC;
#pragma unroll
  for (int i = 0; i < 2; ++i)
#pragma unroll
    for (int j = 0; j < 4; ++j) {
      const int tm = bM + wm * 32 + i * 16;
      const int tn = bN + wn * 64 + j * 16;
#pragma unroll
      for (int r = 0; r < 8; ++r) {
        const int m = tm + r + g * 8;
        const int n = tn + nl;
        float v = acc[i][j][r];
        if (MODE == EPI_F32) {
          ((float*)Cout)[cbase + (long)m * ldc + n] = v;
        } else if (MODE == EPI_BF16) {
          ((__bf16*)Cout)[cbase + (long)m * ldc + n] = (__bf16)v;
        } else if (MODE == EPI_BF16_T) {
          ((__bf16*)Cout)[cbase + (long)n * ldc + m] = (__bf16)v;
        } else if (MODE == EPI_SCORES) {
          v *= scale;
          if (n > m) v = -__builtin_inff();     // causal mask (m,n are in-batch)
          ((float*)Cout)[cbase + (long)m * ldc + n] = v;
        } else { // EPI_BIAS
          ((float*)Cout)[cbase + (long)m * ldc + n] = v + bias[n];
        }
      }
    }
}

// ---------------- row softmax over S=2048, writes f32 (output) + bf16 (for A*V) ----------------

__global__ __launch_bounds__(256)
void softmax_rows(const float* __restrict__ scores, float* __restrict__ Aout,
                  __bf16* __restrict__ Abf) {
  __shared__ float red[256];
  const long row = blockIdx.x;                       // 0 .. B*S-1
  const float* src = scores + row * (long)SEQ;
  float*  dstF = Aout + row * (long)SEQ;
  __bf16* dstB = Abf  + row * (long)SEQ;
  const int t = threadIdx.x;

  float mx = -__builtin_inff();
  for (int k = t; k < SEQ; k += 256) mx = fmaxf(mx, src[k]);
  red[t] = mx; __syncthreads();
  for (int s = 128; s > 0; s >>= 1) { if (t < s) red[t] = fmaxf(red[t], red[t + s]); __syncthreads(); }
  mx = red[0]; __syncthreads();

  float vals[SEQ / 256];
  float sum = 0.f;
#pragma unroll
  for (int i = 0; i < SEQ / 256; ++i) {
    float e = __expf(src[t + i * 256] - mx);   // exp(-inf - mx) = 0 above the diagonal
    vals[i] = e; sum += e;
  }
  red[t] = sum; __syncthreads();
  for (int s = 128; s > 0; s >>= 1) { if (t < s) red[t] += red[t + s]; __syncthreads(); }
  const float inv = 1.0f / red[0];
#pragma unroll
  for (int i = 0; i < SEQ / 256; ++i) {
    float a = vals[i] * inv;
    dstF[t + i * 256] = a;
    dstB[t + i * 256] = (__bf16)a;
  }
}

// ---------------- driver ----------------

extern "C" void kernel_launch(void* const* d_in, const int* in_sizes, int n_in,
                              void* d_out, int out_size, void* d_ws, size_t ws_size,
                              hipStream_t stream) {
  (void)in_sizes; (void)n_in; (void)out_size; (void)ws_size;
  const int*   idx   = (const int*)d_in[0];
  const float* embed = (const float*)d_in[1];
  const float* Wq    = (const float*)d_in[2];
  const float* Wk    = (const float*)d_in[3];
  const float* Wv    = (const float*)d_in[4];
  const float* Wout  = (const float*)d_in[5];
  const float* bout  = (const float*)d_in[6];

  float* logits = (float*)d_out;                                   // [B,S,VOCAB]
  float* Aout   = (float*)d_out + (size_t)BATCH * SEQ * VOCAB;     // [B,S,S]

  char* ws = (char*)d_ws;
  size_t off = 0;
  auto alloc = [&](size_t bytes) -> void* {
    void* p = (void*)(ws + off);
    off += (bytes + 255) & ~(size_t)255;
    return p;
  };
  const size_t MBS = (size_t)BATCH * SEQ;                  // 8192 rows
  __bf16* xbf = (__bf16*)alloc(MBS * DMODEL * 2);
  __bf16* wqb = (__bf16*)alloc((size_t)DMODEL * DMODEL * 2);
  __bf16* wkb = (__bf16*)alloc((size_t)DMODEL * DMODEL * 2);
  __bf16* wvb = (__bf16*)alloc((size_t)DMODEL * DMODEL * 2);
  __bf16* wob = (__bf16*)alloc((size_t)VOCAB * DMODEL * 2);
  __bf16* Qb  = (__bf16*)alloc(MBS * DMODEL * 2);
  __bf16* Kb  = (__bf16*)alloc(MBS * DMODEL * 2);
  __bf16* Vtb = (__bf16*)alloc(MBS * DMODEL * 2);          // [D, B*S] (transposed V)
  float*  sc  = (float*) alloc((size_t)BATCH * SEQ * SEQ * 4);
  __bf16* Ab  = (__bf16*)alloc((size_t)BATCH * SEQ * SEQ * 2);
  __bf16* Ob  = (__bf16*)alloc(MBS * DMODEL * 2);

  const dim3 blk(256);

  cvt_f32_bf16<<<512,  blk, 0, stream>>>(Wq,   wqb, (long)DMODEL * DMODEL);
  cvt_f32_bf16<<<512,  blk, 0, stream>>>(Wk,   wkb, (long)DMODEL * DMODEL);
  cvt_f32_bf16<<<512,  blk, 0, stream>>>(Wv,   wvb, (long)DMODEL * DMODEL);
  cvt_f32_bf16<<<2048, blk, 0, stream>>>(Wout, wob, (long)VOCAB * DMODEL);
  embed_gather<<<(int)MBS, blk, 0, stream>>>(idx, embed, xbf);

  // Q = x Wq^T, K = x Wk^T (bf16);  V stored transposed: Vt[d][m] (bf16)
  const dim3 gqkv(DMODEL / BN, (int)MBS / BM, 1);          // (8, 64)
  gemm_bf16<EPI_BF16><<<gqkv, blk, 0, stream>>>(xbf, 0, DMODEL, wqb, 0, DMODEL,
                                                Qb, 0, DMODEL, DMODEL, nullptr, 0.f);
  gemm_bf16<EPI_BF16><<<gqkv, blk, 0, stream>>>(xbf, 0, DMODEL, wkb, 0, DMODEL,
                                                Kb, 0, DMODEL, DMODEL, nullptr, 0.f);
  gemm_bf16<EPI_BF16_T><<<gqkv, blk, 0, stream>>>(xbf, 0, DMODEL, wvb, 0, DMODEL,
                                                  Vtb, 0, (int)MBS, DMODEL, nullptr, 0.f);

  // scores[b,q,k] = (Q K^T)/sqrt(D) with causal mask -> f32
  const dim3 gsc(SEQ / BN, SEQ / BM, BATCH);               // (16, 16, 4)
  gemm_bf16<EPI_SCORES><<<gsc, blk, 0, stream>>>(Qb, (long)SEQ * DMODEL, DMODEL,
                                                 Kb, (long)SEQ * DMODEL, DMODEL,
                                                 sc, (long)SEQ * SEQ, SEQ,
                                                 DMODEL, nullptr, 1.0f / 32.0f);

  softmax_rows<<<(int)MBS, blk, 0, stream>>>(sc, Aout, Ab);

  // out[b,q,d] = A V : B-operand row n=d is Vt[d][b*S + k], ldb = B*S, batch shift = S columns
  const dim3 gav(DMODEL / BN, SEQ / BM, BATCH);            // (8, 16, 4)
  gemm_bf16<EPI_BF16><<<gav, blk, 0, stream>>>(Ab, (long)SEQ * SEQ, SEQ,
                                               Vtb, (long)SEQ, (int)MBS,
                                               Ob, (long)SEQ * DMODEL, DMODEL,
                                               SEQ, nullptr, 0.f);

  // logits = out Wout^T + bout  (f32 to d_out)
  const dim3 gl(VOCAB / BN, (int)MBS / BM, 1);             // (250, 64)
  gemm_bf16<EPI_BIAS><<<gl, blk, 0, stream>>>(Ob, 0, DMODEL, wob, 0, DMODEL,
                                              logits, 0, VOCAB, DMODEL, bout, 0.f);
}
